// FactorizedDenseAttention_12068858102231
// MI455X (gfx1250) — matrix operand order repaired and checked by reading, not compiled, not run on hardware
//
#include <hip/hip_runtime.h>

typedef __attribute__((ext_vector_type(16))) _Float16 v16h;
typedef __attribute__((ext_vector_type(8)))  _Float16 v8h;
typedef __attribute__((ext_vector_type(8)))  float    v8f;
typedef __attribute__((ext_vector_type(4)))  float    v4f;

#define DIMC   768
#define HEADS  12
#define FDIM   64
#define NBATCH 4
#define SEQ    2048
#define MROWS  (NBATCH*SEQ)   // 8192

// ---------------------------------------------------------------------------
// Kernel 0: weight transpose + f32->f16 convert.  Wt[n][k] = (f16) W[k][n]
// Makes the WMMA B-operand a contiguous 32B load per 16-wide K chunk.
// ---------------------------------------------------------------------------
__global__ void wt_convert_kernel(const float* __restrict__ W,
                                  _Float16* __restrict__ Wt, int K, int N) {
  int idx = blockIdx.x * blockDim.x + threadIdx.x;
  if (idx >= K * N) return;
  int n = idx / K;
  int k = idx - n * K;
  Wt[idx] = (_Float16)W[(size_t)k * N + n];
}

// ---------------------------------------------------------------------------
// Kernel 1: Y[M][N] = X[M][768] * Wt[N][768]  (+ optional bias), f16 WMMA.
// One wave per 16x64 output tile: 4 independent f32 accumulators so the
// converted A fragment is reused 4x per k-step and 4 independent v_wmma
// can pipeline (f16 WMMA->WMMA RAW needs 5 co-exec slots).
// A-fragment layout (16-bit A 16x32): lanes 0-15 -> K {0..7, 16..23},
// lanes 16-31 -> K {8..15, 24..31}; B mirrored with 16-wide K halves.
// ---------------------------------------------------------------------------
__global__ void gemm_wmma_kernel(const float* __restrict__ X,
                                 const _Float16* __restrict__ Wt,
                                 float* __restrict__ Y,
                                 const float* __restrict__ bias, int N) {
  const int K = DIMC;
  int lane = threadIdx.x & 31;
  int wv   = threadIdx.x >> 5;
  int tm   = blockIdx.x * 4 + wv;   // 16-row M tile
  int tn   = blockIdx.y;            // 64-col N tile group
  int l15  = lane & 15;
  int hi   = lane >> 4;             // 0 = lanes 0-15, 1 = lanes 16-31

  int row = tm * 16 + l15;
  const float* xrow = X + (size_t)row * K;
  const _Float16* wrow[4];
#pragma unroll
  for (int d = 0; d < 4; ++d)
    wrow[d] = Wt + (size_t)(tn * 64 + d * 16 + l15) * K;

  v8f acc[4] = {{}, {}, {}, {}};
  for (int k0 = 0; k0 < K; k0 += 32) {
    // Prefetch next A chunk (global_prefetch_b8)
    if (k0 + 32 < K) __builtin_prefetch(xrow + k0 + 32, 0, 1);
    // A fragment: convert 16 f32 -> f16 in-register
    int ka = k0 + hi * 8;
    v4f a0 = *(const v4f*)(xrow + ka);
    v4f a1 = *(const v4f*)(xrow + ka + 4);
    v4f a2 = *(const v4f*)(xrow + ka + 16);
    v4f a3 = *(const v4f*)(xrow + ka + 20);
    v16h af;
#pragma unroll
    for (int i = 0; i < 4; ++i) {
      af[i]      = (_Float16)a0[i];
      af[4 + i]  = (_Float16)a1[i];
      af[8 + i]  = (_Float16)a2[i];
      af[12 + i] = (_Float16)a3[i];
    }
    int kb = k0 + hi * 8;
#pragma unroll
    for (int d = 0; d < 4; ++d) {
      v8h b0 = *(const v8h*)(wrow[d] + kb);
      v8h b1 = *(const v8h*)(wrow[d] + kb + 16);
      v16h bf = __builtin_shufflevector(b0, b1, 0, 1, 2, 3, 4, 5, 6, 7,
                                        8, 9, 10, 11, 12, 13, 14, 15);
      acc[d] = __builtin_amdgcn_wmma_f32_16x16x32_f16(false, af, false, bf,
                                                      (short)0, acc[d],
                                                      false, false);
    }
  }

  if (bias != nullptr) {
#pragma unroll
    for (int d = 0; d < 4; ++d) {
      float bcol = bias[tn * 64 + d * 16 + l15];
#pragma unroll
      for (int r = 0; r < 8; ++r) acc[d][r] += bcol;
    }
  }

  // C/D layout: VGPR r -> row (r + hi*8), col = lane&15
#pragma unroll
  for (int d = 0; d < 4; ++d) {
#pragma unroll
    for (int r = 0; r < 8; ++r) {
      int grow = tm * 16 + r + hi * 8;
      Y[(size_t)grow * N + tn * 64 + d * 16 + l15] = acc[d][r];
    }
  }
}

// ---------------------------------------------------------------------------
// Kernel 2: VbT[b][h][d][k] = (1/32) * sum_{r<32} V[(b*SEQ + 32k + r)][h*64+d]
// Stored k-contiguous (transposed) so the attention B fragment is two
// contiguous 16B loads. (32 consecutive attn columns share one logit.)
// ---------------------------------------------------------------------------
__global__ void vblk_kernel(const float* __restrict__ V,
                            _Float16* __restrict__ VbT) {
  int idx = blockIdx.x * blockDim.x + threadIdx.x;  // < 4*12*64*64
  if (idx >= NBATCH * HEADS * 64 * 64) return;
  int d  = idx & 63;
  int k  = (idx >> 6) & 63;
  int hb = idx >> 12;           // b*HEADS + h
  int h  = hb % HEADS;
  int b  = hb / HEADS;
  const float* vp = V + ((size_t)(b * SEQ + k * 32)) * DIMC + h * FDIM + d;
  float s = 0.f;
#pragma unroll
  for (int r = 0; r < 32; ++r) s += vp[(size_t)r * DIMC];
  VbT[(size_t)hb * 64 * 64 + d * 64 + k] = (_Float16)(s * (1.0f / 32.0f));
}

// ---------------------------------------------------------------------------
// Kernel 3: fused softmax + (w @ Vb) per head.
// Logits per row are only 64 distinct values p[k] = a[k]*b[k>>1].
// Each wave: 16 rows of one (b,h); softmax stats in f32 VALU (__expf),
// then 2 K-steps x 4 D-tiles of v_wmma_f32_16x16x32_f16.
// Output written directly in (B,N,C) layout for the final projection.
// ---------------------------------------------------------------------------
__global__ void attn_kernel(const float* __restrict__ A,
                            const float* __restrict__ Bm,
                            const _Float16* __restrict__ VbT,
                            float* __restrict__ Out) {
  int gwave = (blockIdx.x * blockDim.x + threadIdx.x) >> 5;
  int lane  = threadIdx.x & 31;
  int l15   = lane & 15;
  int hi    = lane >> 4;

  int ntile = gwave & 127;        // SEQ/16 = 128 tiles per (b,h)
  int hb    = gwave >> 7;         // b*HEADS + h
  int h     = hb % HEADS;
  int b     = hb / HEADS;

  int row = b * SEQ + ntile * 16 + l15;
  const float* arow = A  + (size_t)row * DIMC + h * FDIM;          // 64 vals
  const float* brow = Bm + (size_t)row * (HEADS * 32) + h * 32;    // 32 vals

  // Pass 1: row max over the 64 distinct logits
  float mx = -3.4e38f;
#pragma unroll
  for (int k = 0; k < 64; ++k) {
    float p = arow[k] * brow[k >> 1];
    mx = fmaxf(mx, p);
  }
  // Pass 2: denominator (each logit repeats 32x; the 32 is folded into Vb)
  float s = 0.f;
#pragma unroll
  for (int k = 0; k < 64; ++k) s += __expf(arow[k] * brow[k >> 1] - mx);
  float inv = 1.0f / s;

  const _Float16* vb = VbT + (size_t)hb * 64 * 64;  // [d][k], k contiguous
  v8f acc[4] = {{}, {}, {}, {}};
#pragma unroll
  for (int kc = 0; kc < 64; kc += 32) {
    // A fragment = softmax weights in f16
    int ka = kc + hi * 8;
    v16h wf;
#pragma unroll
    for (int i = 0; i < 8; ++i) {
      int k1 = ka + i, k2 = ka + 16 + i;
      wf[i]     = (_Float16)(__expf(arow[k1] * brow[k1 >> 1] - mx) * inv);
      wf[8 + i] = (_Float16)(__expf(arow[k2] * brow[k2 >> 1] - mx) * inv);
    }
    int kb = kc + hi * 8;
#pragma unroll
    for (int d = 0; d < 4; ++d) {
      const _Float16* vcol = vb + (size_t)(d * 16 + l15) * 64 + kb;
      v8h b0 = *(const v8h*)(vcol);
      v8h b1 = *(const v8h*)(vcol + 16);
      v16h bf = __builtin_shufflevector(b0, b1, 0, 1, 2, 3, 4, 5, 6, 7,
                                        8, 9, 10, 11, 12, 13, 14, 15);
      acc[d] = __builtin_amdgcn_wmma_f32_16x16x32_f16(false, wf, false, bf,
                                                      (short)0, acc[d],
                                                      false, false);
    }
  }

#pragma unroll
  for (int d = 0; d < 4; ++d) {
#pragma unroll
    for (int r = 0; r < 8; ++r) {
      int n = ntile * 16 + r + hi * 8;
      Out[((size_t)(b * SEQ + n)) * DIMC + h * FDIM + d * 16 + l15] = acc[d][r];
    }
  }
}

// ---------------------------------------------------------------------------
extern "C" void kernel_launch(void* const* d_in, const int* in_sizes, int n_in,
                              void* d_out, int out_size, void* d_ws,
                              size_t ws_size, hipStream_t stream) {
  (void)in_sizes; (void)n_in; (void)out_size; (void)ws_size;
  const float* x  = (const float*)d_in[0];
  const float* Wa = (const float*)d_in[1];
  const float* Wb = (const float*)d_in[2];
  const float* Wv = (const float*)d_in[3];
  const float* Wp = (const float*)d_in[4];
  const float* bp = (const float*)d_in[5];
  float* out = (float*)d_out;

  char* ws = (char*)d_ws;
  size_t off = 0;
  auto wsalloc = [&](size_t bytes) -> char* {
    char* p = ws + off;
    off += (bytes + 255) & ~(size_t)255;
    return p;
  };
  _Float16* Wta  = (_Float16*)wsalloc((size_t)DIMC * DIMC * 2);
  _Float16* Wtb  = (_Float16*)wsalloc((size_t)DIMC * 384 * 2);
  _Float16* Wtv  = (_Float16*)wsalloc((size_t)DIMC * DIMC * 2);
  _Float16* Wtp  = (_Float16*)wsalloc((size_t)DIMC * DIMC * 2);
  float*    Abuf = (float*)wsalloc((size_t)MROWS * DIMC * 4);
  float*    Bbuf = (float*)wsalloc((size_t)MROWS * 384 * 4);
  float*    Vbuf = (float*)wsalloc((size_t)MROWS * DIMC * 4);
  _Float16* VbT  = (_Float16*)wsalloc((size_t)NBATCH * HEADS * 64 * 64 * 2);
  float*    AttnOut = Vbuf;  // V no longer needed after vblk reduction

  // 0) weight transpose/convert
  wt_convert_kernel<<<(DIMC * DIMC + 255) / 256, 256, 0, stream>>>(Wa, Wta, DIMC, DIMC);
  wt_convert_kernel<<<(DIMC * 384  + 255) / 256, 256, 0, stream>>>(Wb, Wtb, DIMC, 384);
  wt_convert_kernel<<<(DIMC * DIMC + 255) / 256, 256, 0, stream>>>(Wv, Wtv, DIMC, DIMC);
  wt_convert_kernel<<<(DIMC * DIMC + 255) / 256, 256, 0, stream>>>(Wp, Wtp, DIMC, DIMC);

  // 1) projections: a = x@Wa, b = x@Wb, v = x@Wv  (WMMA f16, f32 accum)
  //    grid: (M/16 tiles / 4 waves per block, N/64)
  gemm_wmma_kernel<<<dim3(MROWS / 64, DIMC / 64), 128, 0, stream>>>(x, Wta, Abuf, nullptr, DIMC);
  gemm_wmma_kernel<<<dim3(MROWS / 64, 384  / 64), 128, 0, stream>>>(x, Wtb, Bbuf, nullptr, 384);
  gemm_wmma_kernel<<<dim3(MROWS / 64, DIMC / 64), 128, 0, stream>>>(x, Wtv, Vbuf, nullptr, DIMC);

  // 2) collapse v into 64 block-sums per head (includes /32 of softmax Z)
  vblk_kernel<<<(NBATCH * HEADS * 64 * 64) / 256, 256, 0, stream>>>(Vbuf, VbT);

  // 3) fused softmax + attn@v  (6144 waves total)
  attn_kernel<<<(NBATCH * HEADS * (SEQ / 16) * 32) / 256, 256, 0, stream>>>(Abuf, Bbuf, VbT, AttnOut);

  // 4) final projection with bias, straight into d_out
  gemm_wmma_kernel<<<dim3(MROWS / 64, DIMC / 64), 128, 0, stream>>>(AttnOut, Wtp, out, bp, DIMC);
}
